// MultiheadAttention_39213051412663
// MI455X (gfx1250) — matrix operand order repaired and checked by reading, not compiled, not run on hardware
//
#include <hip/hip_runtime.h>
#include <hip/hip_fp16.h>

typedef _Float16 f16;
typedef __attribute__((ext_vector_type(16))) _Float16 v16h;
typedef __attribute__((ext_vector_type(8)))  _Float16 f16x8;
typedef __attribute__((ext_vector_type(8)))  float    v8f;
typedef __attribute__((ext_vector_type(4)))  int      v4i;

#define S_LEN  2048
#define NBATCH 4
#define DMODEL 1024
#define NHEAD  16
#define DHEAD  64
#define MTOT   (NBATCH * S_LEN)     // 8192 rows
#define LDSK   40                    // 32 + 8 pad (halfs) -> conflict-free b128 LDS loads

#define AS1 __attribute__((address_space(1)))
#define AS3 __attribute__((address_space(3)))

#if defined(__has_builtin)
#if __has_builtin(__builtin_amdgcn_global_load_async_to_lds_b128)
#define USE_ASYNC 1
#else
#define USE_ASYNC 0
#endif
#else
#define USE_ASYNC 0
#endif

template <int N>
__device__ inline void wait_asynccnt() {
#if USE_ASYNC
#if __has_builtin(__builtin_amdgcn_s_wait_asynccnt)
    __builtin_amdgcn_s_wait_asynccnt(N);
#else
    asm volatile("s_wait_asynccnt %0" ::"i"(N) : "memory");
#endif
#endif
}

// ---------- fragment helpers ----------
// A fragment (16x32 f16, M x K), ISA 7.12.2 layout:
// lane<16: row M=lane, halfs = K{0..7, 16..23}; lane>=16: row M=lane-16, K{8..15, 24..31}
__device__ inline v16h load_afrag(const f16* __restrict__ rowbase, int ld, int lane) {
    const int r  = lane & 15;
    const int ko = (lane >> 4) * 8;
    const f16* p = rowbase + (size_t)r * ld + ko;
    f16x8 lo = *(const f16x8*)(p);
    f16x8 hi = *(const f16x8*)(p + 16);
    return __builtin_shufflevector(lo, hi, 0,1,2,3,4,5,6,7,8,9,10,11,12,13,14,15);
}

// B fragment (32x16 f16, K x N) from Bt storage [N][K] (row-major per column):
// lane<16: col N=lane, halfs = K{0..7, 16..23}; lane>=16: col N=lane-16, K{8..15, 24..31}
__device__ inline v16h load_bfrag(const f16* __restrict__ colbase, int ld, int lane) {
    const int n  = lane & 15;
    const int ko = (lane >> 4) * 8;
    const f16* p = colbase + (size_t)n * ld + ko;
    f16x8 lo = *(const f16x8*)(p);
    f16x8 hi = *(const f16x8*)(p + 16);
    return __builtin_shufflevector(lo, hi, 0,1,2,3,4,5,6,7,8,9,10,11,12,13,14,15);
}

// ---------- fp32 -> fp16 convert (x) ----------
__global__ void cvt_f32_f16(const float* __restrict__ src, f16* __restrict__ dst, int n) {
    int i = blockIdx.x * blockDim.x + threadIdx.x;
    int stride = gridDim.x * blockDim.x;
    for (; i < n; i += stride) dst[i] = (f16)src[i];
}

// ---------- fp32 -> fp16 convert + transpose (weights): dst[n][k] = src[k][n] ----------
__global__ __launch_bounds__(256) void cvt_transpose(const float* __restrict__ src,
                                                     f16* __restrict__ dst) {
    __shared__ f16 t[32][33];
    const int bn = blockIdx.x * 32;           // n tile (cols of src)
    const int bk = blockIdx.y * 32;           // k tile (rows of src)
    const int tx = threadIdx.x & 31;
    const int ty = threadIdx.x >> 5;          // 0..7
#pragma unroll
    for (int i = 0; i < 32; i += 8)
        t[ty + i][tx] = (f16)src[(size_t)(bk + ty + i) * DMODEL + bn + tx];
    __syncthreads();
#pragma unroll
    for (int i = 0; i < 32; i += 8)
        dst[(size_t)(bn + ty + i) * DMODEL + bk + tx] = t[tx][ty + i];
}

// ---------- tile staging: 128 rows x 32 halfs, row-major src (stride DMODEL) ----------
__device__ inline void stage_tile(const f16* __restrict__ gbase, f16* lbuf, int tid) {
#pragma unroll
    for (int i = 0; i < 2; ++i) {
        const int c   = tid + i * 256;
        const int row = c >> 2;
        const int off = (c & 3) * 8;
        const f16* gp = gbase + (size_t)row * DMODEL + off;
        f16* lp       = lbuf + row * LDSK + off;
#if USE_ASYNC
        __builtin_amdgcn_global_load_async_to_lds_b128((AS1 v4i*)(f16*)gp,
                                                       (AS3 v4i*)lp, 0, 0);
#else
        *(f16x8*)lp = *(const f16x8*)gp;
#endif
    }
}

// ---------- tiled WMMA GEMM: [MTOT x 1024] x [1024 x 1024] (W pre-transposed) ----------
// MODE 0: QKV projection (blockIdx.z picks weight; epilogue scatters to head layouts)
// MODE 1: output projection, f32 output + bias
template <int MODE>
__global__ __launch_bounds__(256) void gemm_wmma(
    const f16* __restrict__ A,
    const f16* __restrict__ W0, const f16* __restrict__ W1, const f16* __restrict__ W2,
    const float* __restrict__ bias,
    f16* __restrict__ Oq, f16* __restrict__ Ok, f16* __restrict__ Ovt,
    float* __restrict__ Oproj)
{
    __shared__ f16 Asm[2][128 * LDSK];
    __shared__ f16 Bsm[2][128 * LDSK];   // W already transposed: rows are N, cols are K

    const int n0   = blockIdx.x * 128;
    const int m0   = blockIdx.y * 128;
    const int z    = blockIdx.z;
    const f16* W   = (MODE == 1) ? W0 : (z == 0 ? W0 : (z == 1 ? W1 : W2));
    const int tid  = threadIdx.x;
    const int lane = tid & 31;
    const int wave = tid >> 5;
    const int wm   = wave >> 1;   // 0..3 -> 32 rows
    const int wn   = wave & 1;    // 0..1 -> 64 cols

    const f16* Abase = A + (size_t)m0 * DMODEL;
    const f16* Bbase = W + (size_t)n0 * DMODEL;

    v8f acc[2][4];
#pragma unroll
    for (int i = 0; i < 2; ++i)
#pragma unroll
        for (int j = 0; j < 4; ++j)
#pragma unroll
            for (int e = 0; e < 8; ++e) acc[i][j][e] = 0.0f;

    // prologue: async-prefetch tile 0
    stage_tile(Abase, Asm[0], tid);
    stage_tile(Bbase, Bsm[0], tid);

    for (int kt = 0; kt < DMODEL / 32; ++kt) {
        const int cur = kt & 1;
        if (kt + 1 < DMODEL / 32) {
            // prefetch next tile into the other buffer, then wait only for tile kt
            stage_tile(Abase + (kt + 1) * 32, Asm[cur ^ 1], tid);
            stage_tile(Bbase + (kt + 1) * 32, Bsm[cur ^ 1], tid);
            wait_asynccnt<4>();
        } else {
            wait_asynccnt<0>();
        }
        __syncthreads();

        v16h af[2], bf[4];
#pragma unroll
        for (int mi = 0; mi < 2; ++mi)
            af[mi] = load_afrag(Asm[cur] + (wm * 32 + mi * 16) * LDSK, LDSK, lane);
#pragma unroll
        for (int ni = 0; ni < 4; ++ni)
            bf[ni] = load_bfrag(Bsm[cur] + (wn * 64 + ni * 16) * LDSK, LDSK, lane);

#pragma unroll
        for (int mi = 0; mi < 2; ++mi)
#pragma unroll
            for (int ni = 0; ni < 4; ++ni)
                acc[mi][ni] = __builtin_amdgcn_wmma_f32_16x16x32_f16(
                    false, af[mi], false, bf[ni], (short)0, acc[mi][ni], false, false);
        __syncthreads();
    }

    // epilogue: C layout = lane<16 -> M=e, N=lane ; lane>=16 -> M=8+e, N=lane-16
    const int rr = (lane >> 4) * 8;
    const int nl = lane & 15;
#pragma unroll
    for (int mi = 0; mi < 2; ++mi) {
#pragma unroll
        for (int ni = 0; ni < 4; ++ni) {
            const int nglob = n0 + wn * 64 + ni * 16 + nl;
            float bv = 0.0f;
            if constexpr (MODE == 1) bv = bias[nglob];
#pragma unroll
            for (int e = 0; e < 8; ++e) {
                const int mglob = m0 + wm * 32 + mi * 16 + rr + e;
                const float v = acc[mi][ni][e];
                if constexpr (MODE == 1) {
                    Oproj[(size_t)mglob * DMODEL + nglob] = v + bv;
                } else {
                    const int b = mglob >> 11, s = mglob & (S_LEN - 1);
                    const int h = nglob >> 6, d = nglob & (DHEAD - 1);
                    const size_t bh = (size_t)(b * NHEAD + h);
                    if (z == 0)       Oq [(bh * S_LEN + s) * DHEAD + d] = (f16)(v * 0.125f); // fold 1/sqrt(64)
                    else if (z == 1)  Ok [(bh * S_LEN + s) * DHEAD + d] = (f16)v;
                    else              Ovt[(bh * DHEAD + d) * S_LEN + s] = (f16)v;            // V transposed
                }
            }
        }
    }
}

// ---------- causal flash attention, transposed-score formulation ----------
// One wave owns 16 query rows; Sᵀ = K·Qᵀ puts the softmax axis in per-lane registers.
__global__ __launch_bounds__(256) void attn_flash(
    const f16* __restrict__ Q,   // [B*H, S, Dh], pre-scaled by 1/sqrt(Dh)
    const f16* __restrict__ K,   // [B*H, S, Dh]
    const f16* __restrict__ Vt,  // [B*H, Dh, S]
    f16* __restrict__ ctx)       // [B, S, D]
{
    const int bh   = blockIdx.y;
    const int b    = bh >> 4, h = bh & 15;
    const int wave = threadIdx.x >> 5;
    const int lane = threadIdx.x & 31;
    const int q0   = blockIdx.x * 128 + wave * 16;

    const f16* Qb = Q  + (size_t)bh * S_LEN * DHEAD;
    const f16* Kb = K  + (size_t)bh * S_LEN * DHEAD;
    const f16* Vb = Vt + (size_t)bh * DHEAD * S_LEN;

    // Q as B fragments (Bt layout == row-major [q][d]); two d-steps of 32
    v16h qf[2];
#pragma unroll
    for (int ds = 0; ds < 2; ++ds)
        qf[ds] = load_bfrag(Qb + (size_t)q0 * DHEAD + ds * 32, DHEAD, lane);

    float m = -1e30f, l = 0.0f;
    v8f accv[4];
#pragma unroll
    for (int dt = 0; dt < 4; ++dt)
#pragma unroll
        for (int e = 0; e < 8; ++e) accv[dt][e] = 0.0f;

    const int nkb = ((q0 + 15) >> 5) + 1;  // causal: key blocks of 32
    for (int kb = 0; kb < nkb; ++kb) {
        // Sᵀ tiles [16 keys x 16 q], keys split in two 16-tiles, K-dim = Dh in two steps
        v8f st[2];
#pragma unroll
        for (int t = 0; t < 2; ++t) {
#pragma unroll
            for (int e = 0; e < 8; ++e) st[t][e] = 0.0f;
#pragma unroll
            for (int ds = 0; ds < 2; ++ds) {
                v16h kf = load_afrag(Kb + (size_t)(kb * 32 + t * 16) * DHEAD + ds * 32, DHEAD, lane);
                st[t] = __builtin_amdgcn_wmma_f32_16x16x32_f16(
                    false, kf, false, qf[ds], (short)0, st[t], false, false);
            }
        }
        // causal mask (only last block can cross the diagonal); key = row of Sᵀ tile
        if (kb == nkb - 1) {
            const int q = q0 + (lane & 15);
#pragma unroll
            for (int t = 0; t < 2; ++t)
#pragma unroll
                for (int e = 0; e < 8; ++e) {
                    const int key = kb * 32 + t * 16 + ((lane >> 4) * 8) + e;
                    if (key > q) st[t][e] = -1e30f;
                }
        }
        // online softmax: lane holds 16 of 32 keys of its q-column; one xor-16 combine
        float mx = st[0][0];
#pragma unroll
        for (int t = 0; t < 2; ++t)
#pragma unroll
            for (int e = 0; e < 8; ++e) mx = fmaxf(mx, st[t][e]);
        mx = fmaxf(mx, __shfl_xor(mx, 16, 32));
        const float m_new = fmaxf(m, mx);
        const float alpha = __expf(m - m_new);

        float p0[8], p1[8], ls = 0.0f;
#pragma unroll
        for (int e = 0; e < 8; ++e) {
            p0[e] = __expf(st[0][e] - m_new);
            p1[e] = __expf(st[1][e] - m_new);
            ls += p0[e] + p1[e];
        }
        ls += __shfl_xor(ls, 16, 32);
        l = l * alpha + ls;
        m = m_new;

        // repack Pᵀ into B fragment (K=keys 0..31, N=q): lane<16 needs keys {0..7, 16..23},
        // lane>=16 needs keys {8..15, 24..31}; both already held in-lane.
        v16h pf;
#pragma unroll
        for (int e = 0; e < 8; ++e) {
            pf[e]     = (f16)p0[e];
            pf[e + 8] = (f16)p1[e];
        }

        // ctxᵀ = Vᵀ × Pᵀ  (accumulate with rescale)
#pragma unroll
        for (int dt = 0; dt < 4; ++dt) {
#pragma unroll
            for (int e = 0; e < 8; ++e) accv[dt][e] *= alpha;
            v16h vf = load_afrag(Vb + (size_t)(dt * 16) * S_LEN + kb * 32, S_LEN, lane);
            accv[dt] = __builtin_amdgcn_wmma_f32_16x16x32_f16(
                false, vf, false, pf, (short)0, accv[dt], false, false);
        }
    }

    // finalize: ctxᵀ C layout -> contiguous 8-half stores along d
    const float inv = 1.0f / l;
    const int s = q0 + (lane & 15);
#pragma unroll
    for (int dt = 0; dt < 4; ++dt) {
        f16x8 o;
#pragma unroll
        for (int e = 0; e < 8; ++e) o[e] = (f16)(accv[dt][e] * inv);
        *(f16x8*)(ctx + ((size_t)b * S_LEN + s) * DMODEL + h * DHEAD + dt * 16 + (lane >> 4) * 8) = o;
    }
}

extern "C" void kernel_launch(void* const* d_in, const int* in_sizes, int n_in,
                              void* d_out, int out_size, void* d_ws, size_t ws_size,
                              hipStream_t stream) {
    (void)in_sizes; (void)n_in; (void)out_size; (void)ws_size;
    const float* x  = (const float*)d_in[0];
    const float* Wq = (const float*)d_in[1];
    const float* Wk = (const float*)d_in[2];
    const float* Wv = (const float*)d_in[3];
    const float* Wo = (const float*)d_in[4];
    const float* bo = (const float*)d_in[5];

    const size_t NX = (size_t)MTOT * DMODEL;      // 8388608
    const size_t NW = (size_t)DMODEL * DMODEL;    // 1048576
    f16* ws   = (f16*)d_ws;
    f16* xh   = ws;
    f16* wqh  = xh  + NX;   // transposed f16 weights [n][k]
    f16* wkh  = wqh + NW;
    f16* wvh  = wkh + NW;
    f16* woh  = wvh + NW;
    f16* Qb   = woh + NW;
    f16* Kb   = Qb  + NX;
    f16* Vt   = Kb  + NX;
    f16* ctxh = Vt  + NX;

    cvt_f32_f16<<<4096, 256, 0, stream>>>(x, xh, (int)NX);
    dim3 tg(DMODEL / 32, DMODEL / 32);
    cvt_transpose<<<tg, 256, 0, stream>>>(Wq, wqh);
    cvt_transpose<<<tg, 256, 0, stream>>>(Wk, wkh);
    cvt_transpose<<<tg, 256, 0, stream>>>(Wv, wvh);
    cvt_transpose<<<tg, 256, 0, stream>>>(Wo, woh);

    gemm_wmma<0><<<dim3(DMODEL / 128, MTOT / 128, 3), 256, 0, stream>>>(
        xh, wqh, wkh, wvh, nullptr, Qb, Kb, Vt, nullptr);

    attn_flash<<<dim3(S_LEN / 128, NBATCH * NHEAD), 256, 0, stream>>>(Qb, Kb, Vt, ctxh);

    gemm_wmma<1><<<dim3(DMODEL / 128, MTOT / 128, 1), 256, 0, stream>>>(
        ctxh, woh, nullptr, nullptr, bo, nullptr, nullptr, nullptr, (float*)d_out);
}
